// ATT_critic_90718299226433
// MI455X (gfx1250) — matrix operand order repaired and checked by reading, not compiled, not run on hardware
//
#include <hip/hip_runtime.h>

// ---------------------------------------------------------------------------
// Types for CDNA5 WMMA (wave32): v_wmma_f32_16x16x32_bf16
// ---------------------------------------------------------------------------
typedef __attribute__((ext_vector_type(16))) __bf16 bf16x16;
typedef __attribute__((ext_vector_type(8)))  __bf16 bf16x8;
typedef __attribute__((ext_vector_type(8)))  float  floatx8;
typedef int v4i_gcc __attribute__((vector_size(16)));   // matches builtin param type

#define B_ROWS   8192
#define HID      1024
#define HEADS    8
#define ENC_K    544     // 17 * 32
#define DEC_K    96      // 3 * 32

// GEMM block tiling
#define BM       128     // rows per block (8 waves x 16)
#define BN       128     // cols per block (8 WMMA tiles per wave)
#define BK       32      // K-slab per step
#define ROWB     80      // LDS row pitch in bytes (64 data + 16 pad, 16B aligned)

// ---------------------------------------------------------------------------
// gfx1250 async global->LDS copy (guarded; falls back to load+ds_store)
// ---------------------------------------------------------------------------
#if defined(__gfx1250__) && __has_builtin(__builtin_amdgcn_global_load_async_to_lds_b128)
#define HAVE_ASYNC_LDS 1
#else
#define HAVE_ASYNC_LDS 0
#endif

__device__ __forceinline__ void cp_b128_to_lds(const void* g, void* l) {
#if HAVE_ASYNC_LDS
    __builtin_amdgcn_global_load_async_to_lds_b128(
        (v4i_gcc*)g, (v4i_gcc*)l, 0, 0);
#else
    *(bf16x8*)l = *(const bf16x8*)g;
#endif
}

__device__ __forceinline__ void wait_async_copies() {
#if HAVE_ASYNC_LDS
#if __has_builtin(__builtin_amdgcn_s_wait_asynccnt)
    __builtin_amdgcn_s_wait_asynccnt(0);
#else
    asm volatile("s_wait_asynccnt 0x0" ::: "memory");
#endif
#endif
}

// ---------------------------------------------------------------------------
// Pack inputs: enc_in = concat(s, a[:, :32]) -> bf16 [B,544]
//              dec_in = a[:, 32:128]         -> bf16 [B,96]
// ---------------------------------------------------------------------------
__global__ void pack_inputs_kernel(const float* __restrict__ s,
                                   const float* __restrict__ a,
                                   __bf16* __restrict__ enc,
                                   __bf16* __restrict__ dec)
{
    long idx = (long)blockIdx.x * blockDim.x + threadIdx.x;
    long total = (long)B_ROWS * 640;
    if (idx >= total) return;
    int b = (int)(idx / 640);
    int c = (int)(idx % 640);
    if (c < ENC_K) {
        float v = (c < 512) ? s[(long)b * 512 + c]
                            : a[(long)b * 128 + (c - 512)];
        enc[(long)b * ENC_K + c] = (__bf16)v;
    } else {
        int cc = c - ENC_K;
        dec[(long)b * DEC_K + cc] = (__bf16)a[(long)b * 128 + 32 + cc];
    }
}

// ---------------------------------------------------------------------------
// Convert + transpose weights: W[K,N] f32 -> Wt[N,K] bf16 (batched over z)
// ---------------------------------------------------------------------------
__global__ void transpose_bf16_kernel(const float* __restrict__ W,
                                      __bf16* __restrict__ Wt,
                                      int K, int N)
{
    long base = (long)blockIdx.z * K * N;
    long idx  = (long)blockIdx.x * blockDim.x + threadIdx.x;
    if (idx >= (long)K * N) return;
    int n = (int)(idx % N);
    int k = (int)(idx / N);
    Wt[base + (long)n * K + k] = (__bf16)W[base + idx];
}

// ---------------------------------------------------------------------------
// WMMA GEMM: C[M,N] = act(A[M,K] * Bt[N,K]^T + bias), bf16 in/out, f32 acc.
// 256 threads = 8 waves. Block tile 128x128; wave tile 16x128 (8 WMMA/K-step).
// A/B K-slabs (128x32 bf16 each) double-buffered in LDS, filled with
// global_load_async_to_lds_b128; fragments read from LDS with ds_load_b128.
// Fragment layouts per CDNA5 ISA 7.12.2 (wave32):
//   A 16x32 bf16 : lane<16 -> row=lane, K {0..7,16..23}; lane>=16 -> K {8..15,24..31}
//   B 32x16 bf16 : lane<16 -> col=lane, K {0..7,16..23}; lane>=16 -> K {8..15,24..31}
//   C/D 16x16 f32: vgpr j -> row j + 8*(lane>=16), col = lane&15
// Batched over gridDim.z (the 8 attention heads).
// ---------------------------------------------------------------------------
__global__ __launch_bounds__(256)
void gemm_wmma_bf16_kernel(const __bf16* __restrict__ A,
                           const __bf16* __restrict__ Bt,
                           const float* __restrict__ bias,
                           __bf16* __restrict__ C,
                           int M, int N, int K, int relu,
                           long strideBt, long strideBias, long strideC)
{
    Bt   += (long)blockIdx.z * strideBt;
    bias += (long)blockIdx.z * strideBias;
    C    += (long)blockIdx.z * strideC;

    const int nblk = N / BN;
    const long m_blk = (long)(blockIdx.x / nblk) * BM;
    const long n_blk = (long)(blockIdx.x % nblk) * BN;

    __shared__ alignas(16) char smA[2][BM * ROWB];   // 2 x 10240 B
    __shared__ alignas(16) char smB[2][BN * ROWB];   // 2 x 10240 B

    const int tid  = threadIdx.x;
    const int wv   = tid >> 5;
    const int lane = tid & 31;
    const int half = lane >> 4;
    const int row  = lane & 15;

    floatx8 acc[8];
#pragma unroll
    for (int t = 0; t < 8; ++t)
#pragma unroll
        for (int j = 0; j < 8; ++j) acc[t][j] = 0.0f;

    // Stage one 128x32 A slab + 128x32 B slab into LDS buffer `buf`.
    // 512 + 512 b128 chunks; 4 per thread.
    auto stage = [&](int buf, int k0) {
#pragma unroll
        for (int i = 0; i < 2; ++i) {
            int c = tid + i * 256;                // A chunk 0..511
            int r = c >> 2, p = c & 3;
            cp_b128_to_lds(A + (m_blk + r) * (long)K + k0 + p * 8,
                           &smA[buf][r * ROWB + p * 16]);
        }
#pragma unroll
        for (int i = 0; i < 2; ++i) {
            int c = tid + i * 256;                // B chunk 0..511
            int r = c >> 2, p = c & 3;
            cp_b128_to_lds(Bt + (n_blk + r) * (long)K + k0 + p * 8,
                           &smB[buf][r * ROWB + p * 16]);
        }
    };

    stage(0, 0);
    wait_async_copies();
    __syncthreads();

    const int nsteps = K / BK;
    for (int s = 0; s < nsteps; ++s) {
        const int cur = s & 1;
        if (s + 1 < nsteps) stage(cur ^ 1, (s + 1) * BK);   // prefetch next slab

        union { bf16x16 v; bf16x8 h[2]; } af;
        const char* abase = &smA[cur][(wv * 16 + row) * ROWB + half * 16];
        af.h[0] = *(const bf16x8*)(abase);
        af.h[1] = *(const bf16x8*)(abase + 32);
#pragma unroll
        for (int t = 0; t < 8; ++t) {
            union { bf16x16 v; bf16x8 h[2]; } bfr;
            const char* bbase = &smB[cur][(t * 16 + row) * ROWB + half * 16];
            bfr.h[0] = *(const bf16x8*)(bbase);
            bfr.h[1] = *(const bf16x8*)(bbase + 32);
            acc[t] = __builtin_amdgcn_wmma_f32_16x16x32_bf16(
                false, af.v, false, bfr.v, (short)0, acc[t], false, false);
        }

        wait_async_copies();      // next slab landed in LDS
        __syncthreads();          // everyone done reading `cur`
    }

#pragma unroll
    for (int t = 0; t < 8; ++t) {
        int n = (int)n_blk + t * 16 + row;
        float bv = bias[n];
#pragma unroll
        for (int j = 0; j < 8; ++j) {
            long m = m_blk + wv * 16 + half * 8 + j;
            float v = acc[t][j] + bv;
            if (relu) v = fmaxf(v, 0.0f);
            C[m * N + n] = (__bf16)v;
        }
    }
}

// ---------------------------------------------------------------------------
// Attention: one 128-thread block per batch row; each thread owns 8 contiguous
// dims, caches all 8 heads' values in registers (heads read exactly once).
// ---------------------------------------------------------------------------
__global__ __launch_bounds__(128)
void attention_kernel(const __bf16* __restrict__ heads,
                      const __bf16* __restrict__ decH,
                      __bf16* __restrict__ ctx)
{
    int b   = blockIdx.x;
    int tid = threadIdx.x;            // 128 threads, 8 dims each
    int d0  = tid * 8;

    __shared__ float sb[HEADS][128];
    __shared__ float attn[HEADS];

    union { bf16x8 v; __bf16 e[8]; } dv, hv[HEADS];
    dv.v = *(const bf16x8*)(decH + (long)b * HID + d0);

#pragma unroll
    for (int h = 0; h < HEADS; ++h) {
        hv[h].v = *(const bf16x8*)(heads + ((long)h * B_ROWS + b) * HID + d0);
        float sc = 0.0f;
#pragma unroll
        for (int j = 0; j < 8; ++j)
            sc += (float)dv.e[j] * (float)hv[h].e[j];
        sb[h][tid] = sc;
    }
    __syncthreads();

    for (int str = 64; str > 0; str >>= 1) {
        if (tid < str) {
#pragma unroll
            for (int h = 0; h < HEADS; ++h) sb[h][tid] += sb[h][tid + str];
        }
        __syncthreads();
    }

    if (tid == 0) {
        float mx = sb[0][0];
        for (int h = 1; h < HEADS; ++h) mx = fmaxf(mx, sb[h][0]);
        float e[HEADS], sum = 0.0f;
        for (int h = 0; h < HEADS; ++h) { e[h] = __expf(sb[h][0] - mx); sum += e[h]; }
        float inv = 1.0f / sum;
        for (int h = 0; h < HEADS; ++h) attn[h] = e[h] * inv;
    }
    __syncthreads();

    float aw[HEADS];
#pragma unroll
    for (int h = 0; h < HEADS; ++h) aw[h] = attn[h];

    union { bf16x8 v; __bf16 e[8]; } co;
#pragma unroll
    for (int j = 0; j < 8; ++j) {
        float c = 0.0f;
#pragma unroll
        for (int h = 0; h < HEADS; ++h)
            c += aw[h] * (float)hv[h].e[j];
        co.e[j] = (__bf16)c;
    }
    *(bf16x8*)(ctx + (long)b * HID + d0) = co.v;
}

// ---------------------------------------------------------------------------
// fc2: q[b] = <x[b,:], W2[:,0]> + b2   (one wave per row)
// ---------------------------------------------------------------------------
__global__ void fc2_kernel(const __bf16* __restrict__ x,
                           const float* __restrict__ W2,
                           const float* __restrict__ b2,
                           float* __restrict__ q)
{
    int wave = blockIdx.x * (blockDim.x >> 5) + (threadIdx.x >> 5);
    if (wave >= B_ROWS) return;
    int lane = threadIdx.x & 31;
    const __bf16* xr = x + (long)wave * HID;
    float sum = 0.0f;
#pragma unroll
    for (int p = 0; p < 4; ++p) {
        int d = p * 256 + lane * 8;
        union { bf16x8 v; __bf16 e[8]; } xv;
        xv.v = *(const bf16x8*)(xr + d);
#pragma unroll
        for (int j = 0; j < 8; ++j)
            sum += (float)xv.e[j] * W2[d + j];
    }
#pragma unroll
    for (int off = 16; off > 0; off >>= 1)
        sum += __shfl_xor(sum, off, 32);
    if (lane == 0) q[wave] = sum + b2[0];
}

// ---------------------------------------------------------------------------
// Host-side launch
// ---------------------------------------------------------------------------
static inline long cdivl(long a, long b) { return (a + b - 1) / b; }

extern "C" void kernel_launch(void* const* d_in, const int* in_sizes, int n_in,
                              void* d_out, int out_size, void* d_ws, size_t ws_size,
                              hipStream_t stream) {
    (void)in_sizes; (void)n_in; (void)out_size; (void)ws_size;

    const float* s        = (const float*)d_in[0];   // [B, 512]
    const float* a        = (const float*)d_in[1];   // [B, 128]
    const float* W_enc_in = (const float*)d_in[2];   // [544, 1024]
    const float* b_enc_in = (const float*)d_in[3];
    const float* W_dec_in = (const float*)d_in[4];   // [96, 1024]
    const float* b_dec_in = (const float*)d_in[5];
    const float* W_eh     = (const float*)d_in[6];   // [1024, 1024]
    const float* b_eh     = (const float*)d_in[7];
    const float* W_heads  = (const float*)d_in[8];   // [8, 1024, 1024]
    const float* b_heads  = (const float*)d_in[9];   // [8, 1024]
    const float* W_dh     = (const float*)d_in[10];
    const float* b_dh     = (const float*)d_in[11];
    const float* W1       = (const float*)d_in[12];
    const float* b1       = (const float*)d_in[13];
    const float* W2       = (const float*)d_in[14];  // [1024, 1]
    const float* b2       = (const float*)d_in[15];  // [1]
    float* q_out          = (float*)d_out;           // [B, 1]

    char* base = (char*)d_ws;
    size_t off = 0;
    auto carve = [&](size_t bytes) -> char* {
        char* p = base + off;
        off = (off + bytes + 255) & ~(size_t)255;
        return p;
    };
    __bf16* enc_in   = (__bf16*)carve((size_t)B_ROWS * ENC_K * 2);
    __bf16* dec_in   = (__bf16*)carve((size_t)B_ROWS * DEC_K * 2);
    __bf16* Wt_enc   = (__bf16*)carve((size_t)HID * ENC_K * 2);
    __bf16* Wt_dec   = (__bf16*)carve((size_t)HID * DEC_K * 2);
    __bf16* Wt_eh    = (__bf16*)carve((size_t)HID * HID * 2);
    __bf16* Wt_heads = (__bf16*)carve((size_t)HEADS * HID * HID * 2);
    __bf16* Wt_dh    = (__bf16*)carve((size_t)HID * HID * 2);
    __bf16* Wt_1     = (__bf16*)carve((size_t)HID * HID * 2);
    __bf16* enc_lin  = (__bf16*)carve((size_t)B_ROWS * HID * 2);
    __bf16* dec_lin  = (__bf16*)carve((size_t)B_ROWS * HID * 2);
    __bf16* enc_h    = (__bf16*)carve((size_t)B_ROWS * HID * 2);
    __bf16* heads_o  = (__bf16*)carve((size_t)HEADS * B_ROWS * HID * 2); // [H,B,HID]
    __bf16* dec_H    = (__bf16*)carve((size_t)B_ROWS * HID * 2);
    __bf16* ctx      = (__bf16*)carve((size_t)B_ROWS * HID * 2);
    __bf16* x_o      = (__bf16*)carve((size_t)B_ROWS * HID * 2);

    // 1. pack inputs to bf16
    {
        long total = (long)B_ROWS * 640;
        pack_inputs_kernel<<<(unsigned)cdivl(total, 256), 256, 0, stream>>>(
            s, a, enc_in, dec_in);
    }
    // 2. weight transpose+convert
    {
        long n;
        n = (long)ENC_K * HID;
        transpose_bf16_kernel<<<dim3((unsigned)cdivl(n, 256), 1, 1), 256, 0, stream>>>(
            W_enc_in, Wt_enc, ENC_K, HID);
        n = (long)DEC_K * HID;
        transpose_bf16_kernel<<<dim3((unsigned)cdivl(n, 256), 1, 1), 256, 0, stream>>>(
            W_dec_in, Wt_dec, DEC_K, HID);
        n = (long)HID * HID;
        transpose_bf16_kernel<<<dim3((unsigned)cdivl(n, 256), 1, 1), 256, 0, stream>>>(
            W_eh, Wt_eh, HID, HID);
        transpose_bf16_kernel<<<dim3((unsigned)cdivl(n, 256), 1, HEADS), 256, 0, stream>>>(
            W_heads, Wt_heads, HID, HID);
        transpose_bf16_kernel<<<dim3((unsigned)cdivl(n, 256), 1, 1), 256, 0, stream>>>(
            W_dh, Wt_dh, HID, HID);
        transpose_bf16_kernel<<<dim3((unsigned)cdivl(n, 256), 1, 1), 256, 0, stream>>>(
            W1, Wt_1, HID, HID);
    }

    // GEMM launcher: 256 threads; block tile 128x128
    auto gemm = [&](const __bf16* A, const __bf16* Bt, const float* bias,
                    __bf16* C, int M, int N, int K, int relu,
                    int zdim, long sBt, long sBias, long sC) {
        long blocks = (long)(M / BM) * (N / BN);
        gemm_wmma_bf16_kernel<<<dim3((unsigned)blocks, 1, (unsigned)zdim), 256, 0, stream>>>(
            A, Bt, bias, C, M, N, K, relu, sBt, sBias, sC);
    };

    // 3. enc_input = concat @ W_enc_in + b  (linear)
    gemm(enc_in, Wt_enc, b_enc_in, enc_lin, B_ROWS, HID, ENC_K, 0, 1, 0, 0, 0);
    // 4. dec_input = a_others @ W_dec_in + b (linear)
    gemm(dec_in, Wt_dec, b_dec_in, dec_lin, B_ROWS, HID, DEC_K, 0, 1, 0, 0, 0);
    // 5. encoder_h = relu(enc_input @ W_eh + b)
    gemm(enc_lin, Wt_eh, b_eh, enc_h, B_ROWS, HID, HID, 1, 1, 0, 0, 0);
    // 6. encoder_heads[h] = relu(encoder_h @ W_heads[h] + b_heads[h])  (z=8)
    gemm(enc_h, Wt_heads, b_heads, heads_o, B_ROWS, HID, HID, 1, HEADS,
         (long)HID * HID, (long)HID, (long)B_ROWS * HID);
    // 7. decoder_H = relu(dec_input @ W_dh + b)
    gemm(dec_lin, Wt_dh, b_dh, dec_H, B_ROWS, HID, HID, 1, 1, 0, 0, 0);
    // 8. attention
    attention_kernel<<<B_ROWS, 128, 0, stream>>>(heads_o, dec_H, ctx);
    // 9. x = relu(context @ W1 + b1)
    gemm(ctx, Wt_1, b1, x_o, B_ROWS, HID, HID, 1, 1, 0, 0, 0);
    // 10. q = x @ W2 + b2
    fc2_kernel<<<B_ROWS / 8, 256, 0, stream>>>(x_o, W2, b2, q_out);
}
